// GPT2_3384434229601
// MI455X (gfx1250) — compile-verified, matrix operand order checked by reading, not run on hardware
//
#include <hip/hip_runtime.h>
#include <stdint.h>

// GPT-2 forward on gfx1250 (MI455X).
// - All matmuls: v_wmma_f32_16x16x32_f16 (wave32, f32 accumulate).
// - Weights pre-converted fp32->f16 once per launch; activations flow as f16
//   between GEMMs (halves hot HBM/L2 bytes; f16 wte fits easily in 192MB L2).
// - GEMM tiles staged LDS via the Tensor Data Mover (tensor_load_to_lds,
//   TENSORcnt double-buffered), with TDM LDS padding giving the 40-half
//   padded row stride the WMMA fragment loads expect.

#define Bsz 2
#define Tt  1024
#define Vv  50257
#define Dd  768
#define Hh  12
#define Ll  4
#define HDd 64
#define MM  (Bsz * Tt)      // 2048 activation rows

typedef _Float16 h8   __attribute__((ext_vector_type(8)));
typedef _Float16 v16h __attribute__((ext_vector_type(16)));
typedef float    v8f  __attribute__((ext_vector_type(8)));
typedef unsigned int u32x4 __attribute__((ext_vector_type(4)));
typedef unsigned int u32x8 __attribute__((ext_vector_type(8)));

// Build one 16-half WMMA operand fragment from an LDS row (K-contiguous).
// CDNA5 16-bit A/B layout: lanes 0-15 hold K={0..7,16..23}, lanes 16-31 hold
// K={8..15,24..31} -> two contiguous 8-half (16B) ds_load_b128 runs.
__device__ __forceinline__ v16h ld_frag(const _Float16* rowp, int khalf) {
  const h8* p0 = (const h8*)(rowp + 8 * khalf);
  const h8* p1 = (const h8*)(rowp + 16 + 8 * khalf);
  h8 a = *p0, b = *p1;
  return __builtin_shufflevector(a, b, 0,1,2,3,4,5,6,7,8,9,10,11,12,13,14,15);
}

// ---- Tensor Data Mover descriptor helpers (D# groups 0/1, 2D tile) -------
// Group0: count=1 | lds_addr | 57-bit global addr | type=2 ("image").
__device__ __forceinline__ u32x4 tdm_g0(const void* gaddr, unsigned lds_off) {
  unsigned long long a = (unsigned long long)(uintptr_t)gaddr;
  u32x4 g;
  g.x = 1u;                                   // count=1 (valid), no gather
  g.y = lds_off;                              // LDS byte address
  g.z = (unsigned)a;                          // global addr [31:0]
  g.w = (unsigned)((a >> 32) & 0x01FFFFFFull) // global addr [56:32]
        | (2u << 30);                         // type = 2
  return g;
}
// Group1: data_size=2B, pad_enable, pad_interval=16 DW (64B row),
// pad_amount=4 DW (16B) -> LDS row stride 80B = 40 halves.
__device__ __forceinline__ u32x8 tdm_g1(unsigned t_d0, unsigned t_d1,
                                        unsigned tile0, unsigned tile1,
                                        unsigned long long stride0) {
  u32x8 g;
  g[0] = (1u << 16) | (1u << 20) | (3u << 22) | (3u << 25);
  g[1] = (t_d0 & 0xFFFFu) << 16;                       // tensor_dim0 lo
  g[2] = (t_d0 >> 16) | ((t_d1 & 0xFFFFu) << 16);      // dim0 hi | dim1 lo
  g[3] = (t_d1 >> 16) | (tile0 << 16);                 // dim1 hi | tile_dim0
  g[4] = tile1 & 0xFFFFu;                              // tile_dim1 (tile2=0)
  g[5] = (unsigned)stride0;                            // dim0_stride lo
  g[6] = (unsigned)((stride0 >> 32) & 0xFFFFull);      // dim0_stride hi
  g[7] = 0u;
  return g;
}
__device__ __forceinline__ void tdm_load(u32x4 g0, u32x8 g1) {
  asm volatile("tensor_load_to_lds %0, %1" :: "s"(g0), "s"(g1) : "memory");
}

// ---------------------------------------------------------------- convert
__global__ __launch_bounds__(256)
void f32_to_f16_kernel(const float* __restrict__ src,
                       _Float16* __restrict__ dst, int n) {
  int i = blockIdx.x * 256 + threadIdx.x;
  if (i < n) dst[i] = (_Float16)src[i];
}

// ---------------------------------------------------------------- embedding
__global__ __launch_bounds__(256)
void embed_kernel(const int* __restrict__ idx, const float* __restrict__ wte,
                  const float* __restrict__ wpe, float* __restrict__ x) {
  int row = blockIdx.x;            // b*T + t
  int t   = row % Tt;
  int tok = idx[row];
  for (int i = 0; i < 3; ++i) {
    int d = threadIdx.x + i * 256;
    x[(size_t)row * Dd + d] = wte[(size_t)tok * Dd + d] + wpe[(size_t)t * Dd + d];
  }
}

// --------------------------------------------------- layernorm (f32 -> f16)
__global__ __launch_bounds__(256)
void ln_kernel(const float* __restrict__ x, const float* __restrict__ w,
               const float* __restrict__ b, _Float16* __restrict__ out) {
  __shared__ float red[256];
  int row = blockIdx.x, tid = threadIdx.x;
  const float* xr = x + (size_t)row * Dd;
  float v0 = xr[tid], v1 = xr[tid + 256], v2 = xr[tid + 512];
  red[tid] = v0 + v1 + v2;
  __syncthreads();
  for (int k = 128; k > 0; k >>= 1) {
    if (tid < k) red[tid] += red[tid + k];
    __syncthreads();
  }
  float mu = red[0] * (1.0f / Dd);
  __syncthreads();
  float d0 = v0 - mu, d1 = v1 - mu, d2 = v2 - mu;
  red[tid] = d0 * d0 + d1 * d1 + d2 * d2;
  __syncthreads();
  for (int k = 128; k > 0; k >>= 1) {
    if (tid < k) red[tid] += red[tid + k];
    __syncthreads();
  }
  float rs = rsqrtf(red[0] * (1.0f / Dd) + 1e-5f);
  _Float16* o = out + (size_t)row * Dd;
  o[tid]       = (_Float16)(d0 * rs * w[tid]       + b[tid]);
  o[tid + 256] = (_Float16)(d1 * rs * w[tid + 256] + b[tid + 256]);
  o[tid + 512] = (_Float16)(d2 * rs * w[tid + 512] + b[tid + 512]);
}

// ------------------------------- WMMA GEMM with TDM tile staging ----------
// C = A(f16,[M,K]) * W(f16,[N,K])^T (+bias)(+GELU)(+res); C is f32 or f16.
// 128 threads = 4 waves; 64x64 tile, BK=32; double-buffered tensor loads.
template <bool HAS_BIAS, bool HAS_RES, bool DO_GELU, bool OUT_HALF>
__global__ __launch_bounds__(128)
void gemm_kernel(const _Float16* __restrict__ A, const _Float16* __restrict__ W,
                 const float* __restrict__ bias, const float* __restrict__ res,
                 void* __restrict__ Cout, int M, int N, int K) {
  __shared__ alignas(16) _Float16 Ah[2][64 * 40];  // TDM-padded stride 40
  __shared__ alignas(16) _Float16 Bh[2][64 * 40];
  const int tid = threadIdx.x;
  const int lane = tid & 31, wave = tid >> 5;
  const int l16 = lane & 15, khalf = lane >> 4;
  const int m0 = blockIdx.y * 64, n0 = blockIdx.x * 64;

  // LDS byte offsets (generic-pointer low 32 bits == LDS address)
  const unsigned ldsA[2] = {(unsigned)(uintptr_t)&Ah[0][0],
                            (unsigned)(uintptr_t)&Ah[1][0]};
  const unsigned ldsB[2] = {(unsigned)(uintptr_t)&Bh[0][0],
                            (unsigned)(uintptr_t)&Bh[1][0]};
  const u32x8 g1A = tdm_g1(32u, (unsigned)(M - m0), 32u, 64u,
                           (unsigned long long)K);
  const u32x8 g1B = tdm_g1(32u, (unsigned)(N - n0), 32u, 64u,
                           (unsigned long long)K);  // ragged N -> zero fill

  v8f zero8 = {0.f, 0.f, 0.f, 0.f, 0.f, 0.f, 0.f, 0.f};
  v8f acc[4] = {zero8, zero8, zero8, zero8};

  const int nsteps = K / 32;
  if (wave == 0) {   // prologue DMA into buffer 0
    tdm_load(tdm_g0(A + (size_t)m0 * K, ldsA[0]), g1A);
    tdm_load(tdm_g0(W + (size_t)n0 * K, ldsB[0]), g1B);
  }
  for (int s = 0; s < nsteps; ++s) {
    const int cur = s & 1;
    if (wave == 0) {
      if (s + 1 < nsteps) {   // issue next tiles, then wait oldest pair
        tdm_load(tdm_g0(A + (size_t)m0 * K + (size_t)(s + 1) * 32, ldsA[cur ^ 1]), g1A);
        tdm_load(tdm_g0(W + (size_t)n0 * K + (size_t)(s + 1) * 32, ldsB[cur ^ 1]), g1B);
        __builtin_amdgcn_s_wait_tensorcnt(2);
      } else {
        __builtin_amdgcn_s_wait_tensorcnt(0);
      }
    }
    __syncthreads();                       // tiles in `cur` visible to all
    v16h afrag = ld_frag(&Ah[cur][(wave * 16 + l16) * 40], khalf);
    #pragma unroll
    for (int nt = 0; nt < 4; ++nt) {
      v16h bfrag = ld_frag(&Bh[cur][(nt * 16 + l16) * 40], khalf);
      acc[nt] = __builtin_amdgcn_wmma_f32_16x16x32_f16(
          false, afrag, false, bfrag, (short)0, acc[nt], false, false);
    }
    __syncthreads();                       // done reading before overwrite
  }

  // epilogue: C-layout element (m = r + 8*khalf, n = l16) per 16x16 tile
  #pragma unroll
  for (int nt = 0; nt < 4; ++nt) {
    int n = n0 + nt * 16 + l16;
    if (n < N) {
      float bv = HAS_BIAS ? bias[n] : 0.0f;
      #pragma unroll
      for (int r = 0; r < 8; ++r) {
        int m = m0 + wave * 16 + 8 * khalf + r;
        float v = acc[nt][r] + bv;
        if (DO_GELU) v = 0.5f * v * (1.0f + erff(v * 0.70710678118f));
        if (HAS_RES) v += res[(size_t)m * N + n];
        if (OUT_HALF) ((_Float16*)Cout)[(size_t)m * N + n] = (_Float16)v;
        else          ((float*)Cout)[(size_t)m * N + n] = v;
      }
    }
  }
}

// ----------------------------------- causal flash attention (HD=64, T=1024)
// grid: (T/64 query blocks, B*H); 128 threads = 4 waves, 16 queries per wave.
__global__ __launch_bounds__(128)
void attn_kernel(const float* __restrict__ qkv, _Float16* __restrict__ y) {
  __shared__ alignas(16) _Float16 Qh[64 * 72];       // [query][hd]
  __shared__ alignas(16) _Float16 Kh[32 * 72];       // [key][hd]
  __shared__ alignas(16) _Float16 Vt[64 * 40];       // [hd][key] (transposed)
  __shared__ alignas(16) _Float16 Pw[4 * 16 * 40];   // per-wave P tile [q][key]

  const int tid = threadIdx.x;
  const int lane = tid & 31, wave = tid >> 5;
  const int l16 = lane & 15, khalf = lane >> 4;
  const int qb = blockIdx.x;
  const int bb = blockIdx.y / Hh, hh = blockIdx.y % Hh;
  const float scale = 0.125f;   // 1/sqrt(64)

  // stage Q tile 64x64 (f32 -> f16)
  #pragma unroll
  for (int i = 0; i < 8; ++i) {
    int idx = tid + i * 128;
    int r = idx >> 4, c4 = idx & 15;
    float4 v = *(const float4*)(qkv + (size_t)(bb * Tt + qb * 64 + r) * (3 * Dd)
                                + hh * HDd + c4 * 4);
    _Float16* dst = &Qh[r * 72 + c4 * 4];
    dst[0] = (_Float16)v.x; dst[1] = (_Float16)v.y;
    dst[2] = (_Float16)v.z; dst[3] = (_Float16)v.w;
  }

  v8f zero8 = {0.f, 0.f, 0.f, 0.f, 0.f, 0.f, 0.f, 0.f};
  v8f o[4] = {zero8, zero8, zero8, zero8};     // 16 queries x 64 hd
  float run_m[8], run_l[8];
  #pragma unroll
  for (int r = 0; r < 8; ++r) { run_m[r] = -1e30f; run_l[r] = 0.0f; }

  const int jmax = 2 * qb + 1;                 // 32-key blocks up to diagonal
  for (int j = 0; j <= jmax; ++j) {
    if (j < jmax) {  // prefetch next K/V block -> global_prefetch_b8
      const float* nk = qkv + (size_t)(bb * Tt + (j + 1) * 32 + lane) * (3 * Dd)
                        + Dd + hh * HDd;
      __builtin_prefetch(nk, 0, 0);
    }
    __syncthreads();
    // stage K block [32][64] and V block transposed [64 hd][32 key]
    #pragma unroll
    for (int i = 0; i < 4; ++i) {
      int idx = tid + i * 128;
      int r = idx >> 4, c4 = idx & 15;
      size_t rowb = (size_t)(bb * Tt + j * 32 + r) * (3 * Dd) + hh * HDd + c4 * 4;
      float4 kv = *(const float4*)(qkv + rowb + Dd);
      _Float16* kd = &Kh[r * 72 + c4 * 4];
      kd[0] = (_Float16)kv.x; kd[1] = (_Float16)kv.y;
      kd[2] = (_Float16)kv.z; kd[3] = (_Float16)kv.w;
      float4 vv = *(const float4*)(qkv + rowb + 2 * Dd);
      Vt[(c4 * 4 + 0) * 40 + r] = (_Float16)vv.x;
      Vt[(c4 * 4 + 1) * 40 + r] = (_Float16)vv.y;
      Vt[(c4 * 4 + 2) * 40 + r] = (_Float16)vv.z;
      Vt[(c4 * 4 + 3) * 40 + r] = (_Float16)vv.w;
    }
    __syncthreads();

    // S = Q * K^T : 16x32 per wave (2 n-tiles x 2 K-steps over HD=64)
    v8f sc0 = zero8, sc1 = zero8;
    #pragma unroll
    for (int s = 0; s < 2; ++s) {
      v16h aq  = ld_frag(&Qh[(wave * 16 + l16) * 72 + s * 32], khalf);
      v16h bk0 = ld_frag(&Kh[(l16) * 72 + s * 32], khalf);
      sc0 = __builtin_amdgcn_wmma_f32_16x16x32_f16(false, aq, false, bk0,
                                                   (short)0, sc0, false, false);
      v16h bk1 = ld_frag(&Kh[(16 + l16) * 72 + s * 32], khalf);
      sc1 = __builtin_amdgcn_wmma_f32_16x16x32_f16(false, aq, false, bk1,
                                                   (short)0, sc1, false, false);
    }

    // scale + causal mask (C-layout: row m = r + 8*khalf, col = l16)
    const int qrow_base = qb * 64 + wave * 16 + 8 * khalf;
    #pragma unroll
    for (int r = 0; r < 8; ++r) {
      int q = qrow_base + r;
      int k0i = j * 32 + l16;
      float a0 = sc0[r] * scale; if (k0i > q)      a0 = -1e30f;
      float a1 = sc1[r] * scale; if (k0i + 16 > q) a1 = -1e30f;
      sc0[r] = a0; sc1[r] = a1;
    }

    // online softmax: per-row stats across the 16-lane half owning the row
    float fr[8], nm[8];
    #pragma unroll
    for (int r = 0; r < 8; ++r) {
      float mx = fmaxf(sc0[r], sc1[r]);
      mx = fmaxf(mx, __shfl_xor(mx, 1));
      mx = fmaxf(mx, __shfl_xor(mx, 2));
      mx = fmaxf(mx, __shfl_xor(mx, 4));
      mx = fmaxf(mx, __shfl_xor(mx, 8));
      float m2 = fmaxf(run_m[r], mx);
      fr[r] = __expf(run_m[r] - m2);
      nm[r] = m2;
      run_m[r] = m2;
    }
    #pragma unroll
    for (int r = 0; r < 8; ++r) {
      float p0 = __expf(sc0[r] - nm[r]);
      float p1 = __expf(sc1[r] - nm[r]);
      Pw[wave * 640 + (8 * khalf + r) * 40 + l16]      = (_Float16)p0;
      Pw[wave * 640 + (8 * khalf + r) * 40 + 16 + l16] = (_Float16)p1;
      float ls = p0 + p1;
      ls += __shfl_xor(ls, 1);
      ls += __shfl_xor(ls, 2);
      ls += __shfl_xor(ls, 4);
      ls += __shfl_xor(ls, 8);
      run_l[r] = run_l[r] * fr[r] + ls;
    }
    #pragma unroll
    for (int t = 0; t < 4; ++t)
      #pragma unroll
      for (int r = 0; r < 8; ++r) o[t][r] *= fr[r];

    __syncthreads();
    // O += P (16x32) * V (32x64): 4 hd-tiles, one wmma each
    v16h ap = ld_frag(&Pw[wave * 640 + l16 * 40], khalf);
    #pragma unroll
    for (int t = 0; t < 4; ++t) {
      v16h bv = ld_frag(&Vt[(t * 16 + l16) * 40], khalf);
      o[t] = __builtin_amdgcn_wmma_f32_16x16x32_f16(false, ap, false, bv,
                                                    (short)0, o[t], false, false);
    }
  }

  // normalize and write y[b, q, h*64 + hd] as f16 (next GEMM's A operand)
  #pragma unroll
  for (int r = 0; r < 8; ++r) {
    float inv = 1.0f / run_l[r];
    int q = qb * 64 + wave * 16 + 8 * khalf + r;
    _Float16* yr = y + (size_t)(bb * Tt + q) * Dd + hh * HDd;
    #pragma unroll
    for (int t = 0; t < 4; ++t) yr[t * 16 + l16] = (_Float16)(o[t][r] * inv);
  }
}

// ------------------------------------------------------------------ driver
extern "C" void kernel_launch(void* const* d_in, const int* in_sizes, int n_in,
                              void* d_out, int out_size, void* d_ws, size_t ws_size,
                              hipStream_t stream) {
  const int*   idx    = (const int*)d_in[0];
  const float* wte    = (const float*)d_in[1];
  const float* wpe    = (const float*)d_in[2];
  const float* ln1_w  = (const float*)d_in[3];
  const float* ln1_b  = (const float*)d_in[4];
  const float* qkv_w  = (const float*)d_in[5];
  const float* qkv_b  = (const float*)d_in[6];
  const float* proj_w = (const float*)d_in[7];
  const float* proj_b = (const float*)d_in[8];
  const float* ln2_w  = (const float*)d_in[9];
  const float* ln2_b  = (const float*)d_in[10];
  const float* fc_w   = (const float*)d_in[11];
  const float* fc_b   = (const float*)d_in[12];
  const float* fcp_w  = (const float*)d_in[13];
  const float* fcp_b  = (const float*)d_in[14];
  const float* lnf_w  = (const float*)d_in[15];
  const float* lnf_b  = (const float*)d_in[16];
  float* out = (float*)d_out;

  // workspace: f32 {x, qkv} | f16 {h, y, fc} | f16 weight mirrors (~178 MB)
  char* p = (char*)d_ws;
  float*    x      = (float*)p;     p += (size_t)MM * Dd * 4;
  float*    qkvb   = (float*)p;     p += (size_t)MM * 3 * Dd * 4;
  _Float16* hbuf   = (_Float16*)p;  p += (size_t)MM * Dd * 2;
  _Float16* ybuf   = (_Float16*)p;  p += (size_t)MM * Dd * 2;
  _Float16* fcb    = (_Float16*)p;  p += (size_t)MM * 4 * Dd * 2;
  _Float16* qkvw_h = (_Float16*)p;  p += (size_t)Ll * 3 * Dd * Dd * 2;
  _Float16* projw_h= (_Float16*)p;  p += (size_t)Ll * Dd * Dd * 2;
  _Float16* fcw_h  = (_Float16*)p;  p += (size_t)Ll * 4 * Dd * Dd * 2;
  _Float16* fcpw_h = (_Float16*)p;  p += (size_t)Ll * 4 * Dd * Dd * 2;
  _Float16* wte_h  = (_Float16*)p;

  auto conv = [&](const float* s, _Float16* d, size_t n) {
    f32_to_f16_kernel<<<(unsigned)((n + 255) / 256), 256, 0, stream>>>(s, d, (int)n);
  };
  conv(qkv_w,  qkvw_h,  (size_t)Ll * 3 * Dd * Dd);
  conv(proj_w, projw_h, (size_t)Ll * Dd * Dd);
  conv(fc_w,   fcw_h,   (size_t)Ll * 4 * Dd * Dd);
  conv(fcp_w,  fcpw_h,  (size_t)Ll * 4 * Dd * Dd);
  conv(wte,    wte_h,   (size_t)Vv * Dd);

  embed_kernel<<<MM, 256, 0, stream>>>(idx, wte, wpe, x);

  for (int l = 0; l < Ll; ++l) {
    ln_kernel<<<MM, 256, 0, stream>>>(x, ln1_w + l * Dd, ln1_b + l * Dd, hbuf);
    gemm_kernel<true, false, false, false><<<dim3(36, 32), 128, 0, stream>>>(
        hbuf, qkvw_h + (size_t)l * 3 * Dd * Dd, qkv_b + l * 3 * Dd, nullptr,
        qkvb, MM, 3 * Dd, Dd);
    attn_kernel<<<dim3(Tt / 64, Bsz * Hh), 128, 0, stream>>>(qkvb, ybuf);
    gemm_kernel<true, true, false, false><<<dim3(12, 32), 128, 0, stream>>>(
        ybuf, projw_h + (size_t)l * Dd * Dd, proj_b + l * Dd, x, x, MM, Dd, Dd);
    ln_kernel<<<MM, 256, 0, stream>>>(x, ln2_w + l * Dd, ln2_b + l * Dd, hbuf);
    gemm_kernel<true, false, true, true><<<dim3(48, 32), 128, 0, stream>>>(
        hbuf, fcw_h + (size_t)l * 4 * Dd * Dd, fc_b + l * 4 * Dd, nullptr,
        fcb, MM, 4 * Dd, Dd);
    gemm_kernel<true, true, false, false><<<dim3(12, 32), 128, 0, stream>>>(
        fcb, fcpw_h + (size_t)l * Dd * 4 * Dd, fcp_b + l * Dd, x, x, MM, Dd, 4 * Dd);
  }

  ln_kernel<<<MM, 256, 0, stream>>>(x, lnf_w, lnf_b, hbuf);
  // tied lm-head: out[2048, 50257] = h @ wte^T (ragged N zero-filled by TDM)
  gemm_kernel<false, false, false, false><<<dim3((Vv + 63) / 64, 32), 128, 0, stream>>>(
      hbuf, wte_h, nullptr, nullptr, out, MM, Vv, Dd);
}
